// DynaResidualBlockX_34643206210273
// MI455X (gfx1250) — matrix operand order repaired and checked by reading, not compiled, not run on hardware
//
#include <hip/hip_runtime.h>

// ---------------- problem constants ----------------
#define FIN   64
#define FOUT  64
#define FH    128
#define LATD  512
#define NB    16
#define HW    16384          // 128*128 pixels per channel
#define KTOT  53760
#define WELEMS 53248         // total weight elements (before biases)
#define PW    64             // pixels per tile (N dimension, 4 wmma N-tiles)

// weight sub-matrix offsets inside the 53248-element block (row-major [O][I])
#define OFF_IN    0          // 128x64
#define OFF_MIDA  8192       // 128x128
#define OFF_MIDB  24576      // 128x128
#define OFF_OUT   40960      // 64x128
#define OFF_SHORT 49152      // 64x64
// bias offsets inside the 512-element bias block
#define B_IN    0
#define B_MIDA  128
#define B_MIDB  256
#define B_OUT   384
#define B_SHORT 448

#define RS128 0.08838834764831845f   // 1/sqrt(128)
#define RS64  0.125f                 // 1/sqrt(64)

typedef __attribute__((ext_vector_type(16))) __bf16 v16bf;
typedef __attribute__((ext_vector_type(8)))  __bf16 v8bf;
typedef __attribute__((ext_vector_type(8)))  float  v8f;
typedef __attribute__((ext_vector_type(4)))  unsigned int v4ui;
typedef __attribute__((ext_vector_type(8)))  int   v8i;
typedef __attribute__((ext_vector_type(4)))  int   v4i;

// ---------------------------------------------------------------------------
// Stage 1: ks[b][k] = lat[b] . W[k] + bias[k]      (bandwidth-bound on W)
// ---------------------------------------------------------------------------
__global__ __launch_bounds__(256) void hyper_gemm(
    const float* __restrict__ lat,   // [16][512]
    const float* __restrict__ W,     // [53760][512]
    const float* __restrict__ bias,  // [53760]
    float* __restrict__ ks)          // [16][53760]
{
    __shared__ __align__(64) float lat_s[NB * LATD];   // 32 KB
    __shared__ __align__(64) float wtile[256 * 17];    // 17 KB, padded stride

    const int tid = threadIdx.x;
    const int k0  = blockIdx.x * 256;

    for (int i = tid; i < NB * LATD; i += 256) lat_s[i] = lat[i];

    float acc[NB];
#pragma unroll
    for (int b = 0; b < NB; ++b) acc[b] = 0.0f;

    for (int j0 = 0; j0 < LATD; j0 += 16) {
        __syncthreads();
        for (int l = tid; l < 256 * 16; l += 256) {
            int kr = l >> 4, j = l & 15;
            wtile[kr * 17 + j] = W[(size_t)(k0 + kr) * LATD + j0 + j];
        }
        __syncthreads();
#pragma unroll
        for (int j = 0; j < 16; ++j) {
            float wv = wtile[tid * 17 + j];
            const float* lp = lat_s + j0 + j;
#pragma unroll
            for (int b = 0; b < NB; ++b) acc[b] = fmaf(wv, lp[b * LATD], acc[b]);
        }
    }

    const int k  = k0 + tid;
    const float bv = bias[k];
#pragma unroll
    for (int b = 0; b < NB; ++b) ks[(size_t)b * KTOT + k] = acc[b] + bv;
}

// ---------------------------------------------------------------------------
// WMMA fragment builders (CDNA5 wave32 layouts)
// ---------------------------------------------------------------------------
__device__ __forceinline__ v16bf load_afrag(const __bf16* w, int I, int orow0,
                                            int kk, int lane)
{
    const int m   = lane & 15;
    const int klo = (lane < 16) ? 0 : 8;
    const __bf16* p = w + (orow0 + m) * I + kk * 32 + klo;
    v8bf lo = *(const v8bf*)(p);        // K = klo..klo+7
    v8bf hi = *(const v8bf*)(p + 16);   // K = klo+16..klo+23
    v16bf a;
#pragma unroll
    for (int i = 0; i < 8; ++i) { a[i] = lo[i]; a[i + 8] = hi[i]; }
    return a;
}

__device__ __forceinline__ v16bf load_bfrag(const __bf16* act, int kk, int nt,
                                            int lane)
{
    const int kb = kk * 32 + ((lane < 16) ? 0 : 8);
    const int n  = nt * 16 + (lane & 15);
    v16bf b;
#pragma unroll
    for (int i = 0; i < 8; ++i) { b[i] = act[(kb + i) * PW + n]; b[i + 8] = act[(kb + 16 + i) * PW + n]; }
    return b;
}

// B fragment from f32 activations (TDM-staged x tile), converting on the fly
__device__ __forceinline__ v16bf load_bfrag_f32(const float* act, int kk, int nt,
                                                int lane)
{
    const int kb = kk * 32 + ((lane < 16) ? 0 : 8);
    const float* p = act + nt * 16 + (lane & 15);
    v16bf b;
#pragma unroll
    for (int i = 0; i < 8; ++i) { b[i] = (__bf16)p[(kb + i) * PW]; b[i + 8] = (__bf16)p[(kb + 16 + i) * PW]; }
    return b;
}

template <bool F32SRC>
__device__ __forceinline__ void layer(const __bf16* __restrict__ w, int I, int O,
                                      const void* __restrict__ src,
                                      __bf16* __restrict__ dst,
                                      const float* __restrict__ bias,
                                      bool do_relu, int wave, int lane)
{
    const int ntiles = (O >> 4) * 4;
    const int mhi = (lane < 16) ? 0 : 8;
    for (int t = wave; t < ntiles; t += 8) {
        const int ot = t >> 2, nt = t & 3;
        v8f c;
#pragma unroll
        for (int g = 0; g < 8; ++g) c[g] = bias[ot * 16 + g + mhi];
        const int ksteps = I >> 5;
        for (int kk = 0; kk < ksteps; ++kk) {
            v16bf a = load_afrag(w, I, ot * 16, kk, lane);
            v16bf bv;
            if constexpr (F32SRC) bv = load_bfrag_f32((const float*)src, kk, nt, lane);
            else                  bv = load_bfrag((const __bf16*)src, kk, nt, lane);
            c = __builtin_amdgcn_wmma_f32_16x16x32_bf16(
                    false, a, false, bv, (short)0, c, false, false);
        }
        if (do_relu) {
#pragma unroll
            for (int g = 0; g < 8; ++g) c[g] = fmaxf(c[g], 0.0f);
        }
        const int col = nt * 16 + (lane & 15);
#pragma unroll
        for (int g = 0; g < 8; ++g)
            dst[(ot * 16 + g + mhi) * PW + col] = (__bf16)c[g];
    }
}

// ---------------------------------------------------------------------------
// TDM: async 2D tile load (64 channels x 64 pixels, f32) global -> LDS
// D# per CDNA5 ISA ch.8: group0 {count, lds_addr, global_addr, type=2},
// group1 {data_size=4B, tensor_dims, tile 64x64, dim0_stride=HW}
// Toolchain builtin is the 6-arg form: (g0 v4u, g1 v8i, g2 v4i, g3 v4i, g4 v8i, cpol)
// ---------------------------------------------------------------------------
__device__ __forceinline__ void tdm_load_xtile(const float* gptr, unsigned lds_off)
{
    unsigned long long ga = (unsigned long long)(const void*)gptr;
    v4ui g0;
    g0.x = 1u;                                            // count=1, user mode
    g0.y = lds_off;                                       // lds_addr (bytes)
    g0.z = (unsigned)(ga & 0xffffffffu);                  // global_addr[31:0]
    g0.w = (unsigned)((ga >> 32) & 0x01ffffffu)           // global_addr[56:32]
         | (2u << 30);                                    // type = 2 (image)
    v8i g1;
    g1[0] = 0x00020000;            // workgroup_mask=0, data_size=2 (4 bytes)
    g1[1] = (int)0x40000000;       // tensor_dim0 low16 (=16384) in [31:16]
    g1[2] = (int)0x00400000;       // tensor_dim0 hi=0 | tensor_dim1 low16 (=64)
    g1[3] = (int)0x00400000;       // tensor_dim1 hi=0 | tile_dim0 (=64)
    g1[4] = (int)0x00000040;       // tile_dim1 (=64) | tile_dim2=0
    g1[5] = (int)0x00004000;       // tensor_dim0_stride low32 = 16384
    g1[6] = 0;                     // stride hi16 = 0 | dim1_stride low = 0
    g1[7] = 0;
    v4i gz4  = {0, 0, 0, 0};       // groups 2/3 unused (2D tile)
    v8i gz8  = {0, 0, 0, 0, 0, 0, 0, 0};
    __builtin_amdgcn_tensor_load_to_lds(g0, g1, gz4, gz4, gz8, 0);
}

// ---------------------------------------------------------------------------
// Stage 2: per-sample dynamic conv chain, bf16 WMMA, TDM-pipelined x tiles
// grid: 256 blocks (16 samples x 16 pixel-chunks), 256 threads (8 waves)
// ---------------------------------------------------------------------------
__global__ __launch_bounds__(256, 1) void dyna_conv(
    const float* __restrict__ x,    // [16][64][16384]
    const float* __restrict__ ks,   // [16][53760]
    float* __restrict__ out)        // [16][64][16384]
{
    __shared__ __align__(64) __bf16 wlds[WELEMS];      // 106.5 KB weights (bf16, scaled)
    __shared__ __align__(64) float  blds[512];         // conv biases, f32
    __shared__ __align__(64) float  xf[2][FIN * PW];   // 2 x 16 KB f32 x tiles (TDM dst)
    __shared__ __align__(64) __bf16 act_a[FH * PW];    // ping
    __shared__ __align__(64) __bf16 act_b[FH * PW];    // pong

    const int s     = blockIdx.x >> 4;
    const int chunk = blockIdx.x & 15;
    const int tid   = threadIdx.x;
    const int wave  = tid >> 5;
    const int lane  = tid & 31;

    const float* xs = x + (size_t)s * FIN * HW;
    float* os       = out + (size_t)s * FOUT * HW;

    // kick off TDM for tile 0 while we stage the weights
    if (wave == 0) {
        tdm_load_xtile(xs + chunk * 1024,
                       (unsigned)(uintptr_t)(void*)&xf[0][0]);
    }

    // ---- load + scale + convert this sample's hypernet output ----
    const float* kb = ks + (size_t)s * KTOT;
    for (int i = tid; i < WELEMS; i += 256) {
        float v = kb[i] * ((i < OFF_OUT) ? RS128 : RS64);
        wlds[i] = (__bf16)v;
    }
    for (int i = tid; i < 512; i += 256) blds[i] = kb[WELEMS + i];

    const int mhi = (lane < 16) ? 0 : 8;

    for (int pt = 0; pt < 16; ++pt) {
        const int pix0 = chunk * 1024 + pt * PW;
        const float* xt = &xf[pt & 1][0];

        if (wave == 0) {
            if (pt < 15) {   // stream next tile while this one computes
                tdm_load_xtile(xs + pix0 + PW,
                               (unsigned)(uintptr_t)(void*)&xf[(pt + 1) & 1][0]);
                __builtin_amdgcn_s_wait_tensorcnt(1);   // tile pt landed
            } else {
                __builtin_amdgcn_s_wait_tensorcnt(0);
            }
        }
        __syncthreads();   // tile pt visible to all waves (also covers weight staging)

        layer<true >(wlds + OFF_IN,   FIN, FH, xt,    act_a, blds + B_IN,   true, wave, lane);
        __syncthreads();
        layer<false>(wlds + OFF_MIDA, FH,  FH, act_a, act_b, blds + B_MIDA, true, wave, lane);
        __syncthreads();
        layer<false>(wlds + OFF_MIDB, FH,  FH, act_b, act_a, blds + B_MIDB, true, wave, lane);
        __syncthreads();

        // ---- fused final layer: k_out @ h3 + b_out + k_short @ x + b_short ----
        for (int t = wave; t < 16; t += 8) {
            const int ot = t >> 2, nt = t & 3;
            v8f c;
#pragma unroll
            for (int g = 0; g < 8; ++g)
                c[g] = blds[B_OUT + ot * 16 + g + mhi] +
                       blds[B_SHORT + ot * 16 + g + mhi];
            for (int kk = 0; kk < 4; ++kk) {          // k_out: I = 128
                v16bf a  = load_afrag(wlds + OFF_OUT, FH, ot * 16, kk, lane);
                v16bf bv = load_bfrag(act_a, kk, nt, lane);
                c = __builtin_amdgcn_wmma_f32_16x16x32_bf16(
                        false, a, false, bv, (short)0, c, false, false);
            }
            for (int kk = 0; kk < 2; ++kk) {          // k_short: I = 64
                v16bf a  = load_afrag(wlds + OFF_SHORT, FIN, ot * 16, kk, lane);
                v16bf bv = load_bfrag_f32(xt, kk, nt, lane);
                c = __builtin_amdgcn_wmma_f32_16x16x32_bf16(
                        false, a, false, bv, (short)0, c, false, false);
            }
            const int col = pix0 + nt * 16 + (lane & 15);
#pragma unroll
            for (int g = 0; g < 8; ++g)
                os[(ot * 16 + g + mhi) * HW + col] = c[g];
        }
        __syncthreads();   // act_a / xf reuse next tile
    }
}

// ---------------------------------------------------------------------------
extern "C" void kernel_launch(void* const* d_in, const int* in_sizes, int n_in,
                              void* d_out, int out_size, void* d_ws, size_t ws_size,
                              hipStream_t stream)
{
    (void)in_sizes; (void)n_in; (void)out_size; (void)ws_size;
    const float* x    = (const float*)d_in[0];   // [16,64,128,128]
    const float* lat  = (const float*)d_in[1];   // [16,512]
    const float* W    = (const float*)d_in[2];   // [53760,512]
    const float* bias = (const float*)d_in[3];   // [53760]
    float* out = (float*)d_out;                  // [16,64,128,128]
    float* ks  = (float*)d_ws;                   // [16,53760] f32 scratch (3.44 MB)

    hipLaunchKernelGGL(hyper_gemm, dim3(KTOT / 256), dim3(256), 0, stream,
                       lat, W, bias, ks);
    hipLaunchKernelGGL(dyna_conv, dim3(NB * 16), dim3(256), 0, stream,
                       x, ks, out);
}
